// HAMT_4913442587071
// MI455X (gfx1250) — matrix operand order repaired and checked by reading, not compiled, not run on hardware
//
#include <hip/hip_runtime.h>
#include <stdint.h>

// ---------------------------------------------------------------------------
// Problem constants
// ---------------------------------------------------------------------------
#define STEPS   25
#define BS      512
#define HIDDEN  1024
#define IN_DIM  784
#define KPAD1   832          // 784 padded to multiple of 64
#define OUT_DIM 10
#define HL      5
#define ST      (BS * HIDDEN)   // 524288 elements per (batch,hidden) plane

// d_out layout (float offsets)
#define O_SUM  ((size_t)0)                       // 512*10
#define O_S1   ((size_t)5120)                    // 25*512*1024
#define O_S2   ((size_t)(O_S1 + 13107200))
#define O_S3   ((size_t)(O_S2 + 13107200))       // 25*512*10
#define O_STR1 ((size_t)(O_S3 + 128000))
#define O_STR2 ((size_t)(O_STR1 + 524288))

// workspace layout (byte offsets, all 256B aligned)
#define WS_XB    ((size_t)0)             // 12800 x 832 bf16
#define WS_W1B   ((size_t)21299200)      // 1024 x 832 bf16
#define WS_W2B   ((size_t)23003136)      // 1024 x 1024 bf16
#define WS_W3B   ((size_t)25100288)      // 16 x 1024 bf16 (rows 10..15 zero)
#define WS_C1    ((size_t)25133056)      // 25 x 512 x 1024 f32
#define WS_S1B   ((size_t)77561856)      // 512 x 1024 bf16
#define WS_S2B   ((size_t)78610432)      // 512 x 1024 bf16
#define WS_FAM1  ((size_t)79659008)      // ---- zero-init region starts here
#define WS_FAM2  ((size_t)81756160)
#define WS_M1    ((size_t)83853312)
#define WS_M2    ((size_t)85950464)
#define WS_M3    ((size_t)88047616)      // 512 x 16 f32
#define WS_H1    ((size_t)88080384)      // 5 x 512 x 1024 f32
#define WS_H2    ((size_t)98566144)
#define WS_END   ((size_t)109051904)
#define ZERO_FLOATS ((WS_END - WS_FAM1) / 4)     // 7,348,224

typedef __attribute__((ext_vector_type(16))) __bf16 v16bf;
typedef __attribute__((ext_vector_type(8)))  float  v8f;

union Frag { v16bf v; uint4 q[2]; };

// ---------------------------------------------------------------------------
// helpers
// ---------------------------------------------------------------------------
__device__ __forceinline__ float sigmoidf_(float x) {
    return 1.0f / (1.0f + __expf(-x));
}

__device__ __forceinline__ unsigned short f2bf(float f) {
    unsigned int u = __float_as_uint(f);
    unsigned int r = u + 0x7FFFu + ((u >> 16) & 1u);
    if ((u & 0x7F800000u) == 0x7F800000u) r = u;   // inf/nan passthrough
    return (unsigned short)(r >> 16);
}

__device__ __forceinline__ float clamp01(float x) {
    return fminf(fmaxf(x, 0.0f), 1.0f);
}

// ---------------------------------------------------------------------------
// Macro-tiled WMMA block: MT x NT grid of 16x16 tiles per wave.
// A: M x K row-major bf16.  B: N x K row-major bf16 (i.e. operand is B^T).
// K multiple of 32.  MT*NT independent accumulation chains.
// ---------------------------------------------------------------------------
template <int MT, int NT>
__device__ __forceinline__ void wmma_mt(const unsigned short* __restrict__ A,
                                        const unsigned short* __restrict__ B,
                                        int K, int m0, int n0, int lane,
                                        v8f acc[MT * NT]) {
    const int half = lane >> 4;
    const int lr   = lane & 15;
    const unsigned short* ar[MT];
    const unsigned short* br[NT];
#pragma unroll
    for (int mi = 0; mi < MT; mi++)
        ar[mi] = A + (size_t)(m0 + mi * 16 + lr) * K + half * 8;
#pragma unroll
    for (int nj = 0; nj < NT; nj++)
        br[nj] = B + (size_t)(n0 + nj * 16 + lr) * K + half * 8;

    for (int k0 = 0; k0 < K; k0 += 32) {
        Frag a[MT], b[NT];
#pragma unroll
        for (int mi = 0; mi < MT; mi++) {
            a[mi].q[0] = *(const uint4*)(ar[mi] + k0);
            a[mi].q[1] = *(const uint4*)(ar[mi] + k0 + 16);
        }
#pragma unroll
        for (int nj = 0; nj < NT; nj++) {
            b[nj].q[0] = *(const uint4*)(br[nj] + k0);
            b[nj].q[1] = *(const uint4*)(br[nj] + k0 + 16);
        }
#pragma unroll
        for (int mi = 0; mi < MT; mi++)
#pragma unroll
            for (int nj = 0; nj < NT; nj++)
                acc[mi * NT + nj] = __builtin_amdgcn_wmma_f32_16x16x32_bf16(
                    false, a[mi].v, false, b[nj].v, (short)0, acc[mi * NT + nj],
                    false, false);
    }
}

// ---------------------------------------------------------------------------
// utility kernels
// ---------------------------------------------------------------------------
__global__ void k_zero(float* __restrict__ p, size_t n) {
    size_t i = (size_t)blockIdx.x * blockDim.x + threadIdx.x;
    size_t stride = (size_t)gridDim.x * blockDim.x;
    for (; i < n; i += stride) p[i] = 0.0f;
}

// f32 (srcRows x srcK) -> bf16 (dstRows x dstK), zero-padded
__global__ void k_cvt(const float* __restrict__ src, unsigned short* __restrict__ dst,
                      int srcRows, int srcK, int dstRows, int dstK) {
    size_t i = (size_t)blockIdx.x * blockDim.x + threadIdx.x;
    size_t n = (size_t)dstRows * dstK;
    if (i >= n) return;
    int r = (int)(i / dstK);
    int c = (int)(i % dstK);
    float v = (r < srcRows && c < srcK) ? src[(size_t)r * srcK + c] : 0.0f;
    dst[i] = f2bf(v);
}

// ---------------------------------------------------------------------------
// GEMM1 (all steps batched): C = Xb @ W1b^T + b1   (M=12800, N=1024, K=832)
// 32x64 macro-tile per wave (2x4 WMMA tiles) -> 6400 waves, 800 blocks.
// __launch_bounds__(256,1): full VGPR budget, no accumulator spills.
// ---------------------------------------------------------------------------
__global__ void __launch_bounds__(256, 1)
k_gemm1(const unsigned short* __restrict__ A,
        const unsigned short* __restrict__ B,
        const float* __restrict__ bias,
        float* __restrict__ C) {
    int w    = blockIdx.x * 8 + (threadIdx.x >> 5);
    int lane = threadIdx.x & 31;
    int tn = w & 15;     // 16 N-blocks of 64
    int tm = w >> 4;     // 400 M-blocks of 32
    int m0 = tm * 32, n0 = tn * 64;

    v8f acc[8] = {};
    wmma_mt<2, 4>(A, B, KPAD1, m0, n0, lane, acc);

    int half = lane >> 4, lr = lane & 15;
#pragma unroll
    for (int nj = 0; nj < 4; nj++) {
        int n = n0 + nj * 16 + lr;
        float bv = bias[n];
#pragma unroll
        for (int mi = 0; mi < 2; mi++) {
            v8f a = acc[mi * 4 + nj];
#pragma unroll
            for (int i = 0; i < 8; i++) {
                int mrow = m0 + mi * 16 + half * 8 + i;
                C[(size_t)mrow * HIDDEN + n] = a[i] + bv;
            }
        }
    }
}

// ---------------------------------------------------------------------------
// habituation + LIF (layer 1), elementwise over C1[t]
// ---------------------------------------------------------------------------
__global__ void k_hab_leaky(const float* __restrict__ C,
                            float* __restrict__ fam, float* __restrict__ hist,
                            float* __restrict__ m,
                            const float* __restrict__ aP, const float* __restrict__ thP,
                            const float* __restrict__ trP, const float* __restrict__ betaP,
                            float* __restrict__ Sout, unsigned short* __restrict__ sbf,
                            float* __restrict__ strOut, int t) {
    int e = blockIdx.x * 256 + threadIdx.x;
    int j = e & (HIDDEN - 1);
    float xin = C[e];
    float hm = 0.2f * (hist[e] + hist[e + ST] + hist[e + 2 * ST] +
                       hist[e + 3 * ST] + hist[e + 4 * ST]);
    float a  = sigmoidf_(aP[j]);
    float th = sigmoidf_(thP[j]);
    float tr = sigmoidf_(trP[j]);
    float f  = fam[e];
    float denom = fmaxf(fmaxf(fabsf(xin), fabsf(hm)), 1e-8f);
    float nov   = fabsf(xin - hm) / denom;
    float famn  = (nov < 0.3f) ? (th * f + 1.0f - th) : (tr * f);
    famn = clamp01(famn);
    float strn = a * famn;
    float c    = xin * (1.0f - strn);
    fam[e] = famn;
    hist[(size_t)(t % HL) * ST + e] = xin;
    if (t == STEPS - 1) strOut[e] = strn;
    float beta  = clamp01(betaP[0]);
    float mp    = m[e];
    float reset = (mp > 1.0f) ? 1.0f : 0.0f;
    float mn    = fmaf(beta, mp, c) - reset;
    m[e] = mn;
    float s = (mn > 1.0f) ? 1.0f : 0.0f;
    Sout[e] = s;
    sbf[e]  = f2bf(s);
}

// ---------------------------------------------------------------------------
// GEMM2 fused with habituation + LIF (layer 2)
// 32x32 macro-tile per wave (2x2 WMMA tiles) -> 512 waves, 64 blocks.
// ---------------------------------------------------------------------------
__global__ void __launch_bounds__(256, 1)
k_gemm_hab_leaky(const unsigned short* __restrict__ A,
                 const unsigned short* __restrict__ B,
                 const float* __restrict__ bias,
                 const float* __restrict__ aP, const float* __restrict__ thP,
                 const float* __restrict__ trP, const float* __restrict__ betaP,
                 float* __restrict__ fam, float* __restrict__ hist,
                 float* __restrict__ m,
                 float* __restrict__ Sout, unsigned short* __restrict__ sbf,
                 float* __restrict__ strOut, int t) {
    int w    = blockIdx.x * 8 + (threadIdx.x >> 5);
    int lane = threadIdx.x & 31;
    int tn = w & 31;    // 32 N-blocks of 32
    int tm = w >> 5;    // 16 M-blocks of 32
    int m0 = tm * 32, n0 = tn * 32;

    v8f acc[4] = {};
    wmma_mt<2, 2>(A, B, HIDDEN, m0, n0, lane, acc);

    int half = lane >> 4, lr = lane & 15;
    float beta = clamp01(betaP[0]);
    int slot = t % HL;
#pragma unroll
    for (int nj = 0; nj < 2; nj++) {
        int j = n0 + nj * 16 + lr;
        float bv = bias[j];
        float a  = sigmoidf_(aP[j]);
        float th = sigmoidf_(thP[j]);
        float tr = sigmoidf_(trP[j]);
#pragma unroll
        for (int mi = 0; mi < 2; mi++) {
            v8f av = acc[mi * 2 + nj];
#pragma unroll
            for (int i = 0; i < 8; i++) {
                int b = m0 + mi * 16 + half * 8 + i;
                int e = b * HIDDEN + j;
                float xin = av[i] + bv;
                float hm = 0.2f * (hist[e] + hist[e + ST] + hist[e + 2 * ST] +
                                   hist[e + 3 * ST] + hist[e + 4 * ST]);
                float f = fam[e];
                float denom = fmaxf(fmaxf(fabsf(xin), fabsf(hm)), 1e-8f);
                float nov   = fabsf(xin - hm) / denom;
                float famn  = (nov < 0.3f) ? (th * f + 1.0f - th) : (tr * f);
                famn = clamp01(famn);
                float strn = a * famn;
                float c    = xin * (1.0f - strn);
                fam[e] = famn;
                hist[(size_t)slot * ST + e] = xin;
                if (t == STEPS - 1) strOut[e] = strn;
                float mp    = m[e];
                float reset = (mp > 1.0f) ? 1.0f : 0.0f;
                float mn    = fmaf(beta, mp, c) - reset;
                m[e] = mn;
                float s = (mn > 1.0f) ? 1.0f : 0.0f;
                Sout[e] = s;
                sbf[e]  = f2bf(s);
            }
        }
    }
}

// ---------------------------------------------------------------------------
// GEMM3 fused with LIF (output layer, N padded 10->16) + running sum
// ---------------------------------------------------------------------------
__global__ void __launch_bounds__(256, 1)
k_gemm_out(const unsigned short* __restrict__ A,
           const unsigned short* __restrict__ B,
           const float* __restrict__ b3, const float* __restrict__ betaP,
           float* __restrict__ m3,
           float* __restrict__ S3out, float* __restrict__ outsum, int t) {
    int w    = blockIdx.x * 8 + (threadIdx.x >> 5);
    int lane = threadIdx.x & 31;
    v8f acc[1] = {};
    wmma_mt<1, 1>(A, B, HIDDEN, w * 16, 0, lane, acc);
    int half = lane >> 4, lr = lane & 15;
    float bv   = (lr < OUT_DIM) ? b3[lr] : 0.0f;
    float beta = clamp01(betaP[0]);
#pragma unroll
    for (int i = 0; i < 8; i++) {
        int b = w * 16 + half * 8 + i;
        float c3    = acc[0][i] + bv;
        float mp    = m3[b * 16 + lr];
        float reset = (mp > 1.0f) ? 1.0f : 0.0f;
        float mn    = fmaf(beta, mp, c3) - reset;
        m3[b * 16 + lr] = mn;
        float s = (mn > 1.0f) ? 1.0f : 0.0f;
        if (lr < OUT_DIM) {
            S3out[(size_t)t * (BS * OUT_DIM) + b * OUT_DIM + lr] = s;
            outsum[b * OUT_DIM + lr] += s;
        }
    }
}

// ---------------------------------------------------------------------------
// launch
// ---------------------------------------------------------------------------
extern "C" void kernel_launch(void* const* d_in, const int* in_sizes, int n_in,
                              void* d_out, int out_size, void* d_ws, size_t ws_size,
                              hipStream_t stream) {
    (void)in_sizes; (void)n_in; (void)out_size; (void)ws_size;

    const float* x     = (const float*)d_in[0];
    const float* W1    = (const float*)d_in[1];
    const float* b1    = (const float*)d_in[2];
    const float* a1    = (const float*)d_in[3];
    const float* th1   = (const float*)d_in[4];
    const float* tr1   = (const float*)d_in[5];
    const float* beta1 = (const float*)d_in[6];
    const float* W2    = (const float*)d_in[7];
    const float* b2    = (const float*)d_in[8];
    const float* a2    = (const float*)d_in[9];
    const float* th2   = (const float*)d_in[10];
    const float* tr2   = (const float*)d_in[11];
    const float* beta2 = (const float*)d_in[12];
    const float* W3    = (const float*)d_in[13];
    const float* b3    = (const float*)d_in[14];
    const float* beta3 = (const float*)d_in[15];

    char*  ws  = (char*)d_ws;
    float* out = (float*)d_out;

    unsigned short* XB  = (unsigned short*)(ws + WS_XB);
    unsigned short* W1B = (unsigned short*)(ws + WS_W1B);
    unsigned short* W2B = (unsigned short*)(ws + WS_W2B);
    unsigned short* W3B = (unsigned short*)(ws + WS_W3B);
    float*          C1  = (float*)(ws + WS_C1);
    unsigned short* S1B = (unsigned short*)(ws + WS_S1B);
    unsigned short* S2B = (unsigned short*)(ws + WS_S2B);
    float* FAM1 = (float*)(ws + WS_FAM1);
    float* FAM2 = (float*)(ws + WS_FAM2);
    float* M1   = (float*)(ws + WS_M1);
    float* M2   = (float*)(ws + WS_M2);
    float* M3   = (float*)(ws + WS_M3);
    float* H1   = (float*)(ws + WS_H1);
    float* H2   = (float*)(ws + WS_H2);

    // 1. zero recurrent state + output accumulator (every call: deterministic)
    k_zero<<<4096, 256, 0, stream>>>(FAM1, ZERO_FLOATS);
    k_zero<<<32, 256, 0, stream>>>(out + O_SUM, BS * OUT_DIM);

    // 2. bf16 conversions (zero-padded K / rows)
    {
        size_t n = (size_t)STEPS * BS * KPAD1;
        k_cvt<<<(unsigned)((n + 255) / 256), 256, 0, stream>>>(
            x, XB, STEPS * BS, IN_DIM, STEPS * BS, KPAD1);
    }
    k_cvt<<<(HIDDEN * KPAD1 + 255) / 256, 256, 0, stream>>>(
        W1, W1B, HIDDEN, IN_DIM, HIDDEN, KPAD1);
    k_cvt<<<(HIDDEN * HIDDEN + 255) / 256, 256, 0, stream>>>(
        W2, W2B, HIDDEN, HIDDEN, HIDDEN, HIDDEN);
    k_cvt<<<(16 * HIDDEN + 255) / 256, 256, 0, stream>>>(
        W3, W3B, OUT_DIM, HIDDEN, 16, HIDDEN);

    // 3. GEMM1 for all 25 steps at once: (12800 x 832) @ (1024 x 832)^T + b1
    //    400 x 16 macro-tiles (32x64 each) -> 6400 waves -> 800 blocks
    k_gemm1<<<800, 256, 0, stream>>>(XB, W1B, b1, C1);

    // 4. sequential recurrence: only GEMM2/GEMM3 remain inside the t-loop
    for (int t = 0; t < STEPS; t++) {
        k_hab_leaky<<<ST / 256, 256, 0, stream>>>(
            C1 + (size_t)t * ST, FAM1, H1, M1, a1, th1, tr1, beta1,
            out + O_S1 + (size_t)t * ST, S1B, out + O_STR1, t);

        // 16 x 32 macro-tiles (32x32 each) -> 512 waves -> 64 blocks
        k_gemm_hab_leaky<<<64, 256, 0, stream>>>(
            S1B, W2B, b2, a2, th2, tr2, beta2, FAM2, H2, M2,
            out + O_S2 + (size_t)t * ST, S2B, out + O_STR2, t);

        // 32 x 1 tiles -> 32 waves -> 4 blocks
        k_gemm_out<<<4, 256, 0, stream>>>(
            S2B, W3B, b3, beta3, M3, out + O_S3, out + O_SUM, t);
    }
}
